// EGNNNetwork_62809601737238
// MI455X (gfx1250) — matrix operand order repaired and checked by reading, not compiled, not run on hardware
//
#include <hip/hip_runtime.h>
#include <hip/hip_bf16.h>

// ---------------------------------------------------------------------------
// EGNN (N=768, FEAT=HIDDEN=64, 2 layers) for MI455X (gfx1250, wave32, WMMA).
//
// Compute-bound (~39 GFLOP of pair-GEMMs vs ~1MB resident data) -> bf16 WMMA
// (v_wmma_f32_16x16x32_bf16) with f32 accumulation; LN/SiLU epilogues in f32.
//
// Input leaf order assumption (jax tree flatten, dict keys sorted):
//   d_in[0]=x (2304), d_in[1]=h (64),
//   then per layer L (base = 2 + 30*L), nets sorted: net_d, net_e, net_h, net_m;
//   within a net keys sorted: d1, d2, ln1, ln2; dense -> {W,b}; ln -> {bi,s}.
//     net_d: +0 d1.W +1 d1.b +2 d2.W +3 d2.b +4 ln1.bi +5 ln1.s
//     net_e: +6 d1.W +7 d1.b +8 d2.W +9 d2.b +10 ln1.bi +11 ln1.s +12 ln2.bi +13 ln2.s
//     net_h: +14..+21 (same pattern)
//     net_m: +22..+29 (same pattern)
//   d_in[62]=out.W (64), d_in[63]=out.b (1)
// ---------------------------------------------------------------------------

#define NPART 768
#define EPSLN 1e-6f

// LDS strides (odd dword counts to dodge bank conflicts)
#define ZS 65              // f32 row stride of GEMM result buffer
#define AS 66              // bf16 row stride of activation buffer (even: packed u32 loads)
#define ZBYTES (256 * ZS * 4)     // 66560
#define ABYTES (256 * AS * 2)     // 33792
#define WBYTES (4 * 4096 * 2)     // 32768
#define SMEM_TOTAL (ZBYTES + ABYTES + WBYTES + 64*4 + 64*4 + 256*4)  // 134656

typedef __attribute__((ext_vector_type(16))) __bf16 v16bf;
typedef __attribute__((ext_vector_type(8)))  float  v8f;

union Frag { unsigned u[8]; v16bf v; };

__device__ __forceinline__ unsigned short f2bf(float f) {
  union { float f; unsigned u; } x; x.f = f;
  unsigned r = x.u + 0x7FFFu + ((x.u >> 16) & 1u);   // round to nearest even
  return (unsigned short)(r >> 16);
}
__device__ __forceinline__ float bf2f(unsigned short b) {
  union { unsigned u; float f; } x; x.u = ((unsigned)b) << 16;
  return x.f;
}
__device__ __forceinline__ float silu_f(float x) {
  return x / (1.0f + __expf(-x));
}

// ----- WMMA fragment loads (wave32; A and B use the same contiguous k order) -----------

// A: 16x32 bf16 tile. lane = half*16 + (row%16).
// VGPR v: K = 16*half + 2v (+{0,1}).
__device__ __forceinline__ v16bf load_a(const unsigned short* abuf, int rowbase,
                                        int kbase, int ln, int half) {
  Frag f;
  const unsigned short* row = abuf + (rowbase + ln) * AS + kbase;
#pragma unroll
  for (int v = 0; v < 8; ++v) {
    int k = 16 * half + 2 * v;
    f.u[v] = *(const unsigned*)(row + k);
  }
  return f.v;
}

// B: 32x16 bf16 tile from K-contiguous (transposed) weights wt[n][k], k=0..63.
// lanes 0-15: K=0..15 (+kbase), lanes 16-31: K=16..31; N = lane&15.
__device__ __forceinline__ v16bf load_b(const unsigned short* wt, int nbase,
                                        int kbase, int ln, int half) {
  Frag f;
  const unsigned short* col = wt + (nbase + ln) * 64 + kbase + 16 * half;
#pragma unroll
  for (int v = 0; v < 8; ++v)
    f.u[v] = *(const unsigned*)(col + 2 * v);
  return f.v;
}

// Cooperative 256x64 (M x N) GEMM, K=64: abuf(bf16) x wt(bf16, transposed) -> zbuf(f32)
__device__ __forceinline__ void gemm64(const unsigned short* abuf,
                                       const unsigned short* wt, float* zbuf) {
  const int wave = threadIdx.x >> 5;
  const int lane = threadIdx.x & 31;
  const int half = lane >> 4, ln = lane & 15;
#pragma unroll
  for (int mt = 0; mt < 2; ++mt) {
    const int rowbase = wave * 32 + mt * 16;
    v16bf a0 = load_a(abuf, rowbase, 0, ln, half);
    v16bf a1 = load_a(abuf, rowbase, 32, ln, half);
#pragma unroll
    for (int nt = 0; nt < 4; ++nt) {
      v16bf b0 = load_b(wt, nt * 16, 0, ln, half);
      v16bf b1 = load_b(wt, nt * 16, 32, ln, half);
      v8f c = {0.f, 0.f, 0.f, 0.f, 0.f, 0.f, 0.f, 0.f};
      c = __builtin_amdgcn_wmma_f32_16x16x32_bf16(false, a0, false, b0,
                                                  (short)0, c, false, false);
      c = __builtin_amdgcn_wmma_f32_16x16x32_bf16(false, a1, false, b1,
                                                  (short)0, c, false, false);
#pragma unroll
      for (int v = 0; v < 8; ++v)  // D: M = v + 8*half, N = ln
        zbuf[(rowbase + v + 8 * half) * ZS + nt * 16 + ln] = c[v];
    }
  }
}

// Epilogue: bias + LayerNorm + SiLU on own 64-wide row; write bf16 activations
__device__ __forceinline__ void epi_ln_silu(const float* zbuf, unsigned short* abuf,
                                            const float* b, const float* s,
                                            const float* bi) {
  const int r = threadIdx.x;
  const float* zr = zbuf + r * ZS;
  float mu = 0.f;
  for (int n = 0; n < 64; ++n) mu += zr[n] + b[n];
  mu *= (1.f / 64.f);
  float var = 0.f;
  for (int n = 0; n < 64; ++n) { float t = zr[n] + b[n] - mu; var += t * t; }
  var *= (1.f / 64.f);
  const float rs = rsqrtf(var + EPSLN);
  unsigned short* ar = abuf + r * AS;
  for (int n = 0; n < 64; ++n) {
    float t = (zr[n] + b[n] - mu) * rs * s[n] + bi[n];
    ar[n] = f2bf(silu_f(t));
  }
}

// ----- pair kernel ----------------------------------------------------------

struct PairP {
  const float *x, *P1, *P2;
  const float *wr, *b1, *s1e, *bi1e;               // net_e.d1 (row 128) + ln1
  const unsigned short* wtE2; const float *b2e, *s2e, *bi2e;
  const unsigned short* wtM1; const float *bm1, *s1m, *bi1m;
  const unsigned short* wtM2; const float *bm2, *s2m, *bi2m;
  const unsigned short* wtD1; const float *bd1, *s1d, *bi1d;
  const float *wd2, *bd2;
  float *m_i, *dx;
};

__global__ __launch_bounds__(256) void k_pairs(PairP p) {
  extern __shared__ char smem[];
  float*          zbuf = (float*)smem;
  unsigned short* abuf = (unsigned short*)(smem + ZBYTES);
  unsigned short* wlds = (unsigned short*)(smem + ZBYTES + ABYTES);
  float* xi   = (float*)(smem + ZBYTES + ABYTES + WBYTES);
  float* xj   = xi + 64;
  float* darr = xj + 64;

  const int tid = threadIdx.x;
  const int bi0 = blockIdx.x * 16, bj0 = blockIdx.y * 16;

  // Weights -> LDS (u32 copies; all 4 matrices are K-contiguous bf16)
  {
    unsigned* d = (unsigned*)wlds;
    const unsigned* w0 = (const unsigned*)p.wtE2;
    const unsigned* w1 = (const unsigned*)p.wtM1;
    const unsigned* w2 = (const unsigned*)p.wtM2;
    const unsigned* w3 = (const unsigned*)p.wtD1;
    for (int t = tid; t < 2048; t += 256) {
      d[t] = w0[t]; d[2048 + t] = w1[t]; d[4096 + t] = w2[t]; d[6144 + t] = w3[t];
    }
  }
  if (tid < 48)      { int r = tid;      xi[(r/3)*4 + r%3] = p.x[(bi0 + r/3)*3 + r%3]; }
  else if (tid < 96) { int r = tid - 48; xj[(r/3)*4 + r%3] = p.x[(bj0 + r/3)*3 + r%3]; }
  __syncthreads();

  const int il = tid >> 4, jl = tid & 15;
  const int gi = bi0 + il, gj = bj0 + jl;
  const float d0 = xi[il*4+0] - xj[jl*4+0];
  const float d1 = xi[il*4+1] - xj[jl*4+1];
  const float d2 = xi[il*4+2] - xj[jl*4+2];
  const float dist2 = d0*d0 + d1*d1 + d2*d2;

  // S1: z = P1[i] + P2[j] + dist2*W1[128] (bias/LN in epilogue)
  {
    const float* p1r = p.P1 + gi * 64;
    const float* p2r = p.P2 + gj * 64;
    __builtin_prefetch(p1r, 0, 3);
    __builtin_prefetch(p2r, 0, 3);
    float* zr = zbuf + tid * ZS;
    for (int n = 0; n < 64; ++n) zr[n] = p1r[n] + p2r[n] + dist2 * p.wr[n];
  }
  epi_ln_silu(zbuf, abuf, p.b1, p.s1e, p.bi1e);   // own-row only; no barrier needed
  __syncthreads();

  // S2: mass = silu(LN(a1 @ W_e2 + b))
  gemm64(abuf, wlds, zbuf);
  __syncthreads();
  epi_ln_silu(zbuf, abuf, p.b2e, p.s2e, p.bi2e);  // abuf := mass (kept for net_m too)
  __syncthreads();

  // S5: net_d branch first (so abuf=mass survives): mass @ W_d1 -> LN -> silu -> dot w_d2
  gemm64(abuf, wlds + 3 * 4096, zbuf);
  __syncthreads();
  {
    const float* zr = zbuf + tid * ZS;
    float mu = 0.f;
    for (int n = 0; n < 64; ++n) mu += zr[n] + p.bd1[n];
    mu *= (1.f / 64.f);
    float var = 0.f;
    for (int n = 0; n < 64; ++n) { float t = zr[n] + p.bd1[n] - mu; var += t * t; }
    var *= (1.f / 64.f);
    const float rs = rsqrtf(var + EPSLN);
    float acc = 0.f;
    for (int n = 0; n < 64; ++n) {
      float t = (zr[n] + p.bd1[n] - mu) * rs * p.s1d[n] + p.bi1d[n];
      acc += silu_f(t) * p.wd2[n];
    }
    darr[tid] = silu_f(acc + p.bd2[0]);
  }
  __syncthreads();

  // delta_x partial reduction over this j-tile, then one atomic per (i,c)
  if (tid < 48) {
    const int ii = tid / 3, c = tid % 3, gii = bi0 + ii;
    float s = 0.f;
    for (int j = 0; j < 16; ++j) {
      if (gii == bj0 + j) continue;                 // mask diagonal
      s += (xi[ii*4+c] - xj[j*4+c]) * darr[ii * 16 + j];
    }
    atomicAdd(&p.dx[gii * 3 + c], s);
  }

  // S3/S4: m_ij = mlp2(net_m, mass)
  gemm64(abuf, wlds + 1 * 4096, zbuf);
  __syncthreads();
  epi_ln_silu(zbuf, abuf, p.bm1, p.s1m, p.bi1m);
  __syncthreads();
  gemm64(abuf, wlds + 2 * 4096, zbuf);
  __syncthreads();
  {  // final LN+silu written back to zbuf (f32) for the reduction
    float* zr = zbuf + tid * ZS;
    float mu = 0.f;
    for (int n = 0; n < 64; ++n) mu += zr[n] + p.bm2[n];
    mu *= (1.f / 64.f);
    float var = 0.f;
    for (int n = 0; n < 64; ++n) { float t = zr[n] + p.bm2[n] - mu; var += t * t; }
    var *= (1.f / 64.f);
    const float rs = rsqrtf(var + EPSLN);
    for (int n = 0; n < 64; ++n)
      zr[n] = silu_f((zr[n] + p.bm2[n] - mu) * rs * p.s2m[n] + p.bi2m[n]);
  }
  __syncthreads();
  // m_i partial reduction: thread (ii, nq) sums 16 j's over 4 columns
  {
    const int ii = tid >> 4, nq = tid & 15, gii = bi0 + ii;
#pragma unroll
    for (int q = 0; q < 4; ++q) {
      const int n = nq * 4 + q;
      float s = 0.f;
      for (int j = 0; j < 16; ++j) {
        if (gii == bj0 + j) continue;               // mask diagonal
        s += zbuf[(ii * 16 + j) * ZS + n];
      }
      atomicAdd(&p.m_i[gii * 64 + n], s);
    }
  }
}

// ----- small helper kernels -------------------------------------------------

__global__ void k_init(const float* x, const float* h, float* xc, float* hs) {
  int t = blockIdx.x * blockDim.x + threadIdx.x;
  if (t < NPART * 3) xc[t] = x[t];
  if (t < NPART * 64) hs[t] = h[t & 63];
}

__global__ void k_zero(float* mi, float* dx) {
  int t = blockIdx.x * blockDim.x + threadIdx.x;
  if (t < NPART * 64) mi[t] = 0.f;
  if (t < NPART * 3) dx[t] = 0.f;
}

// P1 = Hs @ We1[0:64], P2 = Hs @ We1[64:128]
__global__ void k_prep(const float* hs, const float* We1, float* P1, float* P2) {
  int idx = blockIdx.x * blockDim.x + threadIdx.x;
  if (idx >= NPART * 64) return;
  int r = idx >> 6, n = idx & 63;
  const float* h = hs + r * 64;
  float a = 0.f, b = 0.f;
  for (int k = 0; k < 64; ++k) {
    float hv = h[k];
    a += hv * We1[k * 64 + n];
    b += hv * We1[(64 + k) * 64 + n];
  }
  P1[idx] = a; P2[idx] = b;
}

// Transpose + convert 4 weight matrices to bf16, K-contiguous: wt[m][n*64+k]
__global__ void k_conv(const float* We2, const float* Wm1, const float* Wm2,
                       const float* Wd1, unsigned short* wt) {
  int t = blockIdx.x * blockDim.x + threadIdx.x;
  if (t >= 4 * 4096) return;
  int m = t >> 12, r = t & 4095, n = r >> 6, k = r & 63;
  const float* W = (m == 0) ? We2 : (m == 1) ? Wm1 : (m == 2) ? Wm2 : Wd1;
  wt[m * 4096 + n * 64 + k] = f2bf(W[k * 64 + n]);
}

struct PartP {
  const float *hs_in, *m_i, *Wh1, *bh1, *s1, *bi1, *Wh2, *bh2, *s2, *bi2;
  const float *xc, *dx;
  float *hs_out, *x_out;
};

__global__ __launch_bounds__(64) void k_part(PartP q) {
  __shared__ float hr[64], mr[64], zs[64], aa[64];
  const int i = blockIdx.x, n = threadIdx.x;
  hr[n] = q.hs_in[i * 64 + n];
  mr[n] = q.m_i[i * 64 + n];
  if (n < 3) q.x_out[i * 3 + n] = q.xc[i * 3 + n] + q.dx[i * 3 + n];
  __syncthreads();
  float z = q.bh1[n];
  for (int k = 0; k < 64; ++k)
    z += hr[k] * q.Wh1[k * 64 + n] + mr[k] * q.Wh1[(64 + k) * 64 + n];
  zs[n] = z;
  __syncthreads();
  float mu = 0.f; for (int k = 0; k < 64; ++k) mu += zs[k]; mu *= (1.f / 64.f);
  float var = 0.f; for (int k = 0; k < 64; ++k) { float t = zs[k] - mu; var += t * t; }
  var *= (1.f / 64.f);
  float a = silu_f((z - mu) * rsqrtf(var + EPSLN) * q.s1[n] + q.bi1[n]);
  aa[n] = a;
  __syncthreads();
  float z2 = q.bh2[n];
  for (int k = 0; k < 64; ++k) z2 += aa[k] * q.Wh2[k * 64 + n];
  __syncthreads();
  zs[n] = z2;
  __syncthreads();
  mu = 0.f; for (int k = 0; k < 64; ++k) mu += zs[k]; mu *= (1.f / 64.f);
  var = 0.f; for (int k = 0; k < 64; ++k) { float t = zs[k] - mu; var += t * t; }
  var *= (1.f / 64.f);
  q.hs_out[i * 64 + n] =
      silu_f((z2 - mu) * rsqrtf(var + EPSLN) * q.s2[n] + q.bi2[n]);
}

__global__ void k_final(const float* xc, const float* hs, const float* x0,
                        const float* Wo, const float* bo, float* out) {
  int t = blockIdx.x * blockDim.x + threadIdx.x;
  if (t >= NPART * 3) return;
  out[t] = xc[t] - x0[t];
  int i = t / 3;
  float s = bo[0];
  for (int k = 0; k < 64; ++k) s += hs[i * 64 + k] * Wo[k];
  out[NPART * 3 + t] = s;
}

// ----- host -----------------------------------------------------------------

extern "C" void kernel_launch(void* const* d_in, const int* in_sizes, int n_in,
                              void* d_out, int out_size, void* d_ws, size_t ws_size,
                              hipStream_t stream) {
  const float* x = (const float*)d_in[0];
  const float* h = (const float*)d_in[1];

  float* ws  = (float*)d_ws;
  float* xA  = ws;                  // 2304
  float* xB  = xA + 2304;
  float* hsA = xB + 2304;           // 49152
  float* hsB = hsA + 49152;
  float* P1  = hsB + 49152;
  float* P2  = P1 + 49152;
  float* mi  = P2 + 49152;
  float* dx  = mi + 49152;          // 2304
  unsigned short* wt = (unsigned short*)(dx + 2304);  // 4*4096 bf16

  k_init<<<(49152 + 255) / 256, 256, 0, stream>>>(x, h, xA, hsA);

  float *xc = xA, *xn = xB, *hc = hsA, *hn = hsB;
  for (int L = 0; L < 2; ++L) {
    const int base = 2 + L * 30;
    auto leaf = [&](int o) { return (const float*)d_in[base + o]; };

    k_zero<<<(49152 + 255) / 256, 256, 0, stream>>>(mi, dx);
    k_prep<<<(49152 + 255) / 256, 256, 0, stream>>>(hc, leaf(6), P1, P2);
    k_conv<<<(16384 + 255) / 256, 256, 0, stream>>>(leaf(8), leaf(22), leaf(24),
                                                    leaf(0), wt);
    PairP pp;
    pp.x = xc; pp.P1 = P1; pp.P2 = P2;
    pp.wr = leaf(6) + 128 * 64; pp.b1 = leaf(7); pp.s1e = leaf(11); pp.bi1e = leaf(10);
    pp.wtE2 = wt;             pp.b2e = leaf(9);  pp.s2e = leaf(13); pp.bi2e = leaf(12);
    pp.wtM1 = wt + 4096;      pp.bm1 = leaf(23); pp.s1m = leaf(27); pp.bi1m = leaf(26);
    pp.wtM2 = wt + 8192;      pp.bm2 = leaf(25); pp.s2m = leaf(29); pp.bi2m = leaf(28);
    pp.wtD1 = wt + 12288;     pp.bd1 = leaf(1);  pp.s1d = leaf(5);  pp.bi1d = leaf(4);
    pp.wd2 = leaf(2); pp.bd2 = leaf(3);
    pp.m_i = mi; pp.dx = dx;
    k_pairs<<<dim3(48, 48), 256, SMEM_TOTAL, stream>>>(pp);

    PartP qq;
    qq.hs_in = hc; qq.m_i = mi;
    qq.Wh1 = leaf(14); qq.bh1 = leaf(15); qq.s1 = leaf(19); qq.bi1 = leaf(18);
    qq.Wh2 = leaf(16); qq.bh2 = leaf(17); qq.s2 = leaf(21); qq.bi2 = leaf(20);
    qq.xc = xc; qq.dx = dx; qq.hs_out = hn; qq.x_out = xn;
    k_part<<<NPART, 64, 0, stream>>>(qq);

    float* t;
    t = xc; xc = xn; xn = t;
    t = hc; hc = hn; hn = t;
  }

  k_final<<<(NPART * 3 + 255) / 256, 256, 0, stream>>>(
      xc, hc, x, (const float*)d_in[62], (const float*)d_in[63], (float*)d_out);
}
